// HOAF_22076131901796
// MI455X (gfx1250) — matrix operand order repaired and checked by reading, not compiled, not run on hardware
//
#include <hip/hip_runtime.h>
#include <hip/hip_bf16.h>
#include <stdint.h>

typedef __attribute__((ext_vector_type(16))) _Float16 v16h;
typedef __attribute__((ext_vector_type(8)))  float    v8f;
typedef __attribute__((ext_vector_type(2)))  _Float16 h2;
typedef __attribute__((ext_vector_type(2)))  __fp16   f16x2;

#define HW_ 3136
#define G_  32
#define B_  16
#define KPAD 192
#define WAVES_PER_BLOCK 4
#define SLICES 14
#define TILES_PER_SLICE 7

// ---- original reference index maps ----
__host__ __device__ constexpr int n2o(int b, int c) { return b * 8 - b * (b - 1) / 2 + (c - b); }
__host__ __device__ constexpr int n3o(int a, int b, int c) {
  int off = 0;
  for (int t = 0; t < a; ++t) off += (8 - t) * (9 - t) / 2;
  int m = 8 - a, bb = b - a;
  return off + bb * m - bb * (bb - 1) / 2 + (c - b);
}

// ---- packed-math feature permutation: new column k -> (weight tensor, flat index) ----
struct PermT { short deg[KPAD]; short idx[KPAD]; };
constexpr PermT make_perm() {
  PermT p{};
  int k = 0;
  for (int i = 0; i < 8; ++i) { p.deg[k] = 0; p.idx[k] = (short)i; ++k; }            // deg1
  for (int i = 0; i < 8; ++i)                                                         // d2 full slots
    for (int q = (i + 1) >> 1; q < 4; ++q)
      for (int sel = 0; sel < 2; ++sel) { p.deg[k] = 1; p.idx[k] = (short)n2o(i, 2 * q + sel); ++k; }
  for (int i = 1; i < 8; i += 2) { p.deg[k] = 1; p.idx[k] = (short)n2o(i, i); ++k; }  // d2 diagonals
  for (int a = 0; a < 8; ++a)                                                         // d3 main
    for (int i = 0; i < 8; ++i)
      for (int q = (i + 1) >> 1; q < 4; ++q)
        if (i >= a)
          for (int sel = 0; sel < 2; ++sel) { p.deg[k] = 2; p.idx[k] = (short)n3o(a, i, 2 * q + sel); ++k; }
  for (int i = 1; i < 8; i += 2)                                                      // d3 diagonals
    for (int aa = 0; aa <= i; ++aa) { p.deg[k] = 2; p.idx[k] = (short)n3o(aa, i, i); ++k; }
  for (; k < KPAD; ++k) { p.deg[k] = 3; p.idx[k] = 0; }                               // zero pad
  return p;
}
__device__ __constant__ PermT c_perm = make_perm();

__device__ __forceinline__ uint32_t h2bits(h2 v) {
  union { h2 h; uint32_t u; } un; un.h = v; return un.u;
}
__device__ __forceinline__ h2 bcast(h2 v, int sel) {
  _Float16 s = sel ? v[1] : v[0];
  h2 r; r[0] = s; r[1] = s; return r;
}
__device__ __forceinline__ h2 cvt_pk(float a, float b) {
  f16x2 r = __builtin_amdgcn_cvt_pkrtz(a, b);
  union { f16x2 f; h2 h; } un; un.f = r; return un.h;
}

// Pack permuted [w1|w2|w3] -> f16 Wcat[G][16][192] (rows 8..15 zero), bsum = b1+b2+b3
__global__ void hoaf_prep(const float* __restrict__ w1, const float* __restrict__ w2,
                          const float* __restrict__ w3, const float* __restrict__ b1,
                          const float* __restrict__ b2, const float* __restrict__ b3,
                          _Float16* __restrict__ wc, float* __restrict__ bs) {
  int idx = blockIdx.x * blockDim.x + threadIdx.x;
  if (idx < G_ * 16 * KPAD) {
    int g = idx / (16 * KPAD);
    int r = (idx / KPAD) & 15;
    int k = idx % KPAD;
    float v = 0.f;
    if (r < 8) {
      int d = c_perm.deg[k], id = c_perm.idx[k];
      if (d == 0)      v = w1[(g * 8 + r) * 8 + id];
      else if (d == 1) v = w2[(g * 8 + r) * 36 + id];
      else if (d == 2) v = w3[(g * 8 + r) * 120 + id];
    }
    wc[idx] = (_Float16)v;
  }
  if (idx < G_ * 8) bs[idx] = b1[idx] + b2[idx] + b3[idx];
}

__global__ void __launch_bounds__(128) hoaf_main(const float* __restrict__ x,
                                                 const _Float16* __restrict__ wc,
                                                 const float* __restrict__ bs,
                                                 float* __restrict__ out) {
  // per-wave feature tile: 32 pixels x 100 dwords (192 halves used + pad), 16B-aligned rows
  __shared__ uint32_t lds[WAVES_PER_BLOCK][32][100];

  const int wave  = threadIdx.x >> 5;
  const int lane  = threadIdx.x & 31;
  const int wid   = blockIdx.x * WAVES_PER_BLOCK + wave;
  const int bg    = wid / SLICES;      // (b*32 + g)
  const int slice = wid % SLICES;
  const int g     = bg & (G_ - 1);

  uint32_t* myrow = &lds[wave][lane][0];
  #pragma unroll
  for (int q = 82; q < 96; ++q) myrow[q] = 0u;   // zero K = 164..191 pad once

  // ---- A fragments (resident): 16-bit A 16x32 layout
  const _Float16* wrow = wc + ((size_t)g * 16 + (lane & 15)) * KPAD;
  const int hsel = (lane & 16) ? 16 : 0;
  v16h afrag[6];
  #pragma unroll
  for (int kk = 0; kk < 6; ++kk) {
    union { uint4 u[2]; v16h h; } un;
    un.u[0] = *(const uint4*)(wrow + kk * 32 + hsel);
    un.u[1] = *(const uint4*)(wrow + kk * 32 + hsel + 8);
    afrag[kk] = un.h;
  }

  float bias[8];
  #pragma unroll
  for (int o = 0; o < 8; ++o) bias[o] = bs[g * 8 + o];

  const uint32_t* brow0 = &lds[wave][lane & 15][0];
  const uint32_t* brow1 = &lds[wave][16 + (lane & 15)][0];
  const int bsel = (lane & 16) ? 8 : 0;

  for (int t = 0; t < TILES_PER_SLICE; ++t) {
    const int pixBase = (slice * TILES_PER_SLICE + t) * 32;

    const float* xp32 = x + (size_t)bg * 8 * HW_ + pixBase + lane;
    float xv[8];
    #pragma unroll
    for (int k = 0; k < 8; ++k) xv[k] = xp32[(size_t)k * HW_];

    // ---- packed-f16 feature expansion (order matches c_perm) ----
    h2 xp[4];
    #pragma unroll
    for (int q = 0; q < 4; ++q) xp[q] = cvt_pk(xv[2 * q], xv[2 * q + 1]);
    #pragma unroll
    for (int q = 0; q < 4; ++q) myrow[q] = h2bits(xp[q]);          // fi 0..7

    h2 d2s[16];
    {
      int u = 4, s = 0;
      #pragma unroll
      for (int i = 0; i < 8; ++i) {
        h2 bi = bcast(xp[i >> 1], i & 1);
        #pragma unroll
        for (int q = (i + 1) >> 1; q < 4; ++q) {
          h2 v = bi * xp[q];
          d2s[s] = v; myrow[u] = h2bits(v); ++s; ++u;              // fi 8..39
        }
      }
      h2 o13; o13[0] = xp[0][1]; o13[1] = xp[1][1];
      h2 o57; o57[0] = xp[2][1]; o57[1] = xp[3][1];
      h2 L1 = o13 * o13, L2 = o57 * o57;
      myrow[20] = h2bits(L1); myrow[21] = h2bits(L2);              // fi 40..43

      u = 22;                                                      // d3 main: fi 44..143
      #pragma unroll
      for (int a = 0; a < 8; ++a) {
        h2 ba = bcast(xp[a >> 1], a & 1);
        int s2 = 0;
        #pragma unroll
        for (int i = 0; i < 8; ++i) {
          #pragma unroll
          for (int q = (i + 1) >> 1; q < 4; ++q) {
            if (i >= a) { myrow[u] = h2bits(ba * d2s[s2]); ++u; }
            ++s2;
          }
        }
      }
      // d3 diagonals: fi 144..163 (u = 72..81)
      h2 diag[4] = { bcast(L1, 0), bcast(L1, 1), bcast(L2, 0), bcast(L2, 1) };
      #pragma unroll
      for (int ii = 0; ii < 4; ++ii) {
        const int imax = 2 * ii + 1;
        #pragma unroll
        for (int pp = 0; pp <= (imax - 1) / 2; ++pp) { myrow[u] = h2bits(xp[pp] * diag[ii]); ++u; }
      }
    }

    asm volatile("s_wait_dscnt 0" ::: "memory");   // intra-wave cross-lane LDS handoff

    // ---- GEMM: 6 K-chunks x 2 pixel sub-tiles, f32 accumulate
    v8f c0 = {}; v8f c1 = {};
    #pragma unroll
    for (int kk = 0; kk < 6; ++kk) {
      union { uint4 u[2]; v16h h; } ub0, ub1;
      ub0.u[0] = *(const uint4*)(brow0 + kk * 16 + bsel);
      ub0.u[1] = *(const uint4*)(brow0 + kk * 16 + bsel + 4);
      ub1.u[0] = *(const uint4*)(brow1 + kk * 16 + bsel);
      ub1.u[1] = *(const uint4*)(brow1 + kk * 16 + bsel + 4);
      c0 = __builtin_amdgcn_wmma_f32_16x16x32_f16(false, afrag[kk], false, ub0.h,
                                                  (short)0, c0, false, false);
      c1 = __builtin_amdgcn_wmma_f32_16x16x32_f16(false, afrag[kk], false, ub1.h,
                                                  (short)0, c1, false, false);
    }

    asm volatile("s_wait_dscnt 0" ::: "memory");

    // ---- merge sub-tile 1 into lanes 16..31, full 128B coalesced stores
    #pragma unroll
    for (int o = 0; o < 8; ++o) {
      float hi = __shfl(c1[o], lane & 15, 32);
      float v  = ((lane & 16) ? hi : c0[o]) + bias[o];
      out[((size_t)bg * 8 + o) * HW_ + pixBase + lane] = v;
    }
  }
}

extern "C" void kernel_launch(void* const* d_in, const int* in_sizes, int n_in,
                              void* d_out, int out_size, void* d_ws, size_t ws_size,
                              hipStream_t stream) {
  (void)in_sizes; (void)n_in; (void)out_size; (void)ws_size;
  const float* x  = (const float*)d_in[0];
  const float* w1 = (const float*)d_in[1];
  const float* b1 = (const float*)d_in[2];
  const float* w2 = (const float*)d_in[3];
  const float* b2 = (const float*)d_in[4];
  const float* w3 = (const float*)d_in[5];
  const float* b3 = (const float*)d_in[6];
  float* out = (float*)d_out;

  _Float16* wcat = (_Float16*)d_ws;
  float* bsum = (float*)((char*)d_ws + (size_t)G_ * 16 * KPAD * sizeof(_Float16));

  hoaf_prep<<<(G_ * 16 * KPAD + 255) / 256, 256, 0, stream>>>(w1, w2, w3, b1, b2, b3, wcat, bsum);

  const int totalWaves = B_ * G_ * SLICES;               // 7168 waves, 7 tiles each
  hoaf_main<<<totalWaves / WAVES_PER_BLOCK, 32 * WAVES_PER_BLOCK, 0, stream>>>(x, wcat, bsum, out);
}
